// JITRotaryAttention_15272903704644
// MI455X (gfx1250) — matrix operand order repaired and checked by reading, not compiled, not run on hardware
//
#include <hip/hip_runtime.h>
#include <math.h>

// ---------------------------------------------------------------------------
// CDNA5 / gfx1250 rotary attention: all matmuls via v_wmma_f32_16x16x32_bf16
// ---------------------------------------------------------------------------

typedef __attribute__((ext_vector_type(16))) __bf16 v16bf;
typedef __attribute__((ext_vector_type(8)))  float  v8f;

union FragU { unsigned u[8]; v16bf v; };

__device__ __forceinline__ __bf16 to_bf16(float f) {
  union { float f; unsigned u; } in; in.f = f;
  unsigned r = (in.u + 0x7FFFu + ((in.u >> 16) & 1u)) >> 16;  // RNE
  union { unsigned short s; __bf16 b; } out;
  out.s = (unsigned short)r;
  return out.b;
}

__device__ __forceinline__ float bf16_to_f32(__bf16 b) {
  union { __bf16 b; unsigned short s; } in; in.b = b;
  union { unsigned u; float f; } out; out.u = ((unsigned)in.s) << 16;
  return out.f;
}

__device__ __forceinline__ v8f wmma_bf16(v16bf a, v16bf b, v8f c) {
  return __builtin_amdgcn_wmma_f32_16x16x32_bf16(false, a, false, b,
                                                 (short)0, c, false, false);
}

// A-matrix fragment, bf16 16x32 (MxK). ISA 7.12.2 layout:
//  lanes 0-15: M=lane,  V0..3 = K{0,1..7},  V4..7 = K{16..23}
//  lanes16-31: M=lane-16,V0..3 = K{8..15},  V4..7 = K{24..31}
__device__ __forceinline__ v16bf load_fragA(const __bf16* tile, int lane, int ld) {
  const int row = lane & 15;
  const int hi8 = (lane >> 4) << 3;
  const unsigned short* p = (const unsigned short*)tile + (size_t)row * ld;
  FragU f;
#pragma unroll
  for (int v = 0; v < 8; ++v) {
    const int k = (v < 4) ? (hi8 + 2 * v) : (16 + hi8 + 2 * (v - 4));
    f.u[v] = *(const unsigned*)(p + k);
  }
  return f.v;
}

// B-matrix fragment, bf16 32x16 (KxN), source stored row-major [N, ld] so that
// element (k, n) = src[n*ld + k].  Layout (sparse-doc analog):
//  lanes 0-15: N=lane,  V0..3 = K{0..7},  V4..7 = K{16..23}
//  lanes16-31: N=lane-16,V0..3 = K{8..15}, V4..7 = K{24..31}
__device__ __forceinline__ v16bf load_fragB(const __bf16* tile, int lane, int ld) {
  const int n  = lane & 15;
  const int kh = (lane >> 4) << 3;
  const unsigned short* p = (const unsigned short*)tile + (size_t)n * ld;
  FragU f;
#pragma unroll
  for (int v = 0; v < 8; ++v) f.u[v] = *(const unsigned*)(p + ((v < 4) ? (kh + 2 * v) : (16 + kh + 2 * (v - 4))));
  return f.v;
}

// ---------------------------------------------------------------------------
// Kernel 1: fp32 -> bf16 conversion
// ---------------------------------------------------------------------------
__global__ void f32_to_bf16_kernel(const float* __restrict__ in,
                                   __bf16* __restrict__ out, int n) {
  int i = blockIdx.x * blockDim.x + threadIdx.x;
  if (i < n) out[i] = to_bf16(in[i]);
}

// ---------------------------------------------------------------------------
// Kernel 2/5: C[M,N] = A[M,K] * W[N,K]^T (+bias). Per-wave 32x64 tile:
// each B fragment is loaded once and feeds two WMMAs (two M sub-tiles),
// 8 WMMAs per 12 b128 fragment loads. 4 waves / block -> 256 N per block.
// ---------------------------------------------------------------------------
template <bool BF16_OUT>
__global__ void __launch_bounds__(128)
gemm_bf16_wmma(const __bf16* __restrict__ A, const __bf16* __restrict__ W,
               const float* __restrict__ bias, void* __restrict__ Cout,
               int N, int K) {
  const int lane  = threadIdx.x & 31;
  const int wave  = threadIdx.x >> 5;
  const int mbase = blockIdx.x * 32;
  const int n0    = (blockIdx.y * 4 + wave) * 64;
  const int hi8   = (lane >> 4) << 3;
  const int nl    = lane & 15;

  const v8f z = {0.f,0.f,0.f,0.f,0.f,0.f,0.f,0.f};
  v8f accA0 = z, accA1 = z, accA2 = z, accA3 = z;   // rows mbase..mbase+15
  v8f accB0 = z, accB1 = z, accB2 = z, accB3 = z;   // rows mbase+16..mbase+31

  const __bf16* arow0 = A + (size_t)mbase * K;
  const __bf16* arow1 = arow0 + (size_t)16 * K;
  for (int k = 0; k < K; k += 32) {
    __builtin_prefetch(arow0 + k + 256, 0, 1);  // global_prefetch upcoming A
    const v16bf a0 = load_fragA(arow0 + k, lane, K);
    const v16bf a1 = load_fragA(arow1 + k, lane, K);
    v16bf b;
    b = load_fragB(W + (size_t)(n0     ) * K + k, lane, K);
    accA0 = wmma_bf16(a0, b, accA0); accB0 = wmma_bf16(a1, b, accB0);
    b = load_fragB(W + (size_t)(n0 + 16) * K + k, lane, K);
    accA1 = wmma_bf16(a0, b, accA1); accB1 = wmma_bf16(a1, b, accB1);
    b = load_fragB(W + (size_t)(n0 + 32) * K + k, lane, K);
    accA2 = wmma_bf16(a0, b, accA2); accB2 = wmma_bf16(a1, b, accB2);
    b = load_fragB(W + (size_t)(n0 + 48) * K + k, lane, K);
    accA3 = wmma_bf16(a0, b, accA3); accB3 = wmma_bf16(a1, b, accB3);
  }

#pragma unroll
  for (int t = 0; t < 4; ++t) {
    const v8f aT = (t == 0) ? accA0 : (t == 1) ? accA1 : (t == 2) ? accA2 : accA3;
    const v8f bT = (t == 0) ? accB0 : (t == 1) ? accB1 : (t == 2) ? accB2 : accB3;
    const int n = n0 + t * 16 + nl;
    const float bv = bias ? bias[n] : 0.0f;
#pragma unroll
    for (int r = 0; r < 8; ++r) {
      const size_t i0 = (size_t)(mbase + hi8 + r) * N + n;
      const size_t i1 = (size_t)(mbase + 16 + hi8 + r) * N + n;
      if constexpr (BF16_OUT) {
        ((__bf16*)Cout)[i0] = to_bf16(aT[r] + bv);
        ((__bf16*)Cout)[i1] = to_bf16(bT[r] + bv);
      } else {
        ((float*)Cout)[i0] = aT[r] + bv;
        ((float*)Cout)[i1] = bT[r] + bv;
      }
    }
  }
}

// ---------------------------------------------------------------------------
// Kernel 3: RoPE on q,k + repack.  qkv (bf16) layout [B*S, 3*D],
// e = t*D + h*64 + d.  Outputs: qb,kb [bh, S, 64] bf16; v transposed
// vbT [bh, 64, S] bf16 so PV B-fragments get contiguous (k,k+1) dwords.
// One thread per (b,h,s,i), i in [0,32) handles the rotation pair (i, i+32).
// ---------------------------------------------------------------------------
__global__ void rope_pack_kernel(const __bf16* __restrict__ qkv,
                                 __bf16* __restrict__ qb,
                                 __bf16* __restrict__ kb,
                                 __bf16* __restrict__ vbT) {
  constexpr int S = 2048, H = 16, HD = 64, D = 1024;
  const int idx = blockIdx.x * blockDim.x + threadIdx.x;
  const int i = idx & 31;
  int t = idx >> 5;
  const int s = t & (S - 1); t >>= 11;
  const int h = t & (H - 1);
  const int b = t >> 4;
  if (b >= 2) return;

  const size_t row  = (size_t)(b * S + s) * (3 * D);
  const size_t qoff = row + h * HD + i;
  const float q1 = bf16_to_f32(qkv[qoff]);         const float q2 = bf16_to_f32(qkv[qoff + 32]);
  const float k1 = bf16_to_f32(qkv[qoff + D]);     const float k2 = bf16_to_f32(qkv[qoff + D + 32]);
  const float v1 = bf16_to_f32(qkv[qoff + 2 * D]); const float v2 = bf16_to_f32(qkv[qoff + 2 * D + 32]);

  // inv_freq = 10000^(-2i/64)
  const float inv_freq = __expf(-(float)(2 * i) * (9.2103403720f / 64.0f));
  const float ang = (float)s * inv_freq;
  float sn, cs;
  __sincosf(ang, &sn, &cs);

  const size_t bh = (size_t)(b * H + h);
  __bf16* qd = qb + (bh * S + s) * HD;
  __bf16* kd = kb + (bh * S + s) * HD;
  qd[i]      = to_bf16(q1 * cs - q2 * sn);
  qd[i + 32] = to_bf16(q2 * cs + q1 * sn);
  kd[i]      = to_bf16(k1 * cs - k2 * sn);
  kd[i + 32] = to_bf16(k2 * cs + k1 * sn);

  __bf16* vd = vbT + bh * (size_t)HD * S;
  vd[(size_t)i * S + s]        = to_bf16(v1);
  vd[(size_t)(i + 32) * S + s] = to_bf16(v2);
}

// ---------------------------------------------------------------------------
// Kernel 4: flash attention.  Block = 4 waves; wave w owns 16 q-rows.
// Per 32-key step: 4 WMMA for S = Q K^T, online softmax (half-wave
// shfl_xor reductions, wave32), P staged in per-wave LDS (C-layout ->
// A-layout), 4 WMMA for O += P V.  Output bf16 [B*S, D].
// ---------------------------------------------------------------------------
__global__ void __launch_bounds__(128)
attn_kernel(const __bf16* __restrict__ qb, const __bf16* __restrict__ kb,
            const __bf16* __restrict__ vbT, __bf16* __restrict__ aout) {
  constexpr int S = 2048, H = 16, HD = 64;
  __shared__ __align__(16) __bf16 pshare[4][16 * 32];

  const int lane  = threadIdx.x & 31;
  const int wave  = threadIdx.x >> 5;
  const int bh    = blockIdx.y;                    // b*H + h
  const int qbase = blockIdx.x * 64 + wave * 16;
  const int hi8   = (lane >> 4) << 3;
  const int nl    = lane & 15;

  const __bf16* qptr = qb  + ((size_t)bh * S + qbase) * HD;
  const __bf16* kptr = kb  + (size_t)bh * S * HD;
  const __bf16* vptr = vbT + (size_t)bh * HD * S;

  const v16bf qf0 = load_fragA(qptr,      lane, HD);   // dims 0..31
  const v16bf qf1 = load_fragA(qptr + 32, lane, HD);   // dims 32..63

  float mrow[8], lrow[8];
  v8f o0 = {0.f,0.f,0.f,0.f,0.f,0.f,0.f,0.f};
  v8f o1 = o0, o2 = o0, o3 = o0;
#pragma unroll
  for (int r = 0; r < 8; ++r) { mrow[r] = -3.0e38f; lrow[r] = 0.f; }

  __bf16* ldsw = pshare[wave];
  const float scale = 0.125f;  // hd^-0.5 = 1/8

  for (int kb0 = 0; kb0 < S; kb0 += 32) {
    v8f s0 = {0.f,0.f,0.f,0.f,0.f,0.f,0.f,0.f};
    v8f s1 = s0;
    {
      v16bf kf;
      kf = load_fragB(kptr + (size_t)kb0 * HD,             lane, HD);
      s0 = wmma_bf16(qf0, kf, s0);
      kf = load_fragB(kptr + (size_t)kb0 * HD + 32,        lane, HD);
      s0 = wmma_bf16(qf1, kf, s0);
      kf = load_fragB(kptr + (size_t)(kb0 + 16) * HD,      lane, HD);
      s1 = wmma_bf16(qf0, kf, s1);
      kf = load_fragB(kptr + (size_t)(kb0 + 16) * HD + 32, lane, HD);
      s1 = wmma_bf16(qf1, kf, s1);
    }

#pragma unroll
    for (int r = 0; r < 8; ++r) {
      const float a = s0[r] * scale;
      const float b = s1[r] * scale;
      float mx = fmaxf(a, b);
      mx = fmaxf(mx, __shfl_xor(mx, 1, 32));
      mx = fmaxf(mx, __shfl_xor(mx, 2, 32));
      mx = fmaxf(mx, __shfl_xor(mx, 4, 32));
      mx = fmaxf(mx, __shfl_xor(mx, 8, 32));     // row max across 16 lanes of half
      const float mnew  = fmaxf(mrow[r], mx);
      const float alpha = __expf(mrow[r] - mnew);
      mrow[r] = mnew;
      const float pa = __expf(a - mnew);
      const float pb = __expf(b - mnew);
      float rs = pa + pb;
      rs += __shfl_xor(rs, 1, 32);
      rs += __shfl_xor(rs, 2, 32);
      rs += __shfl_xor(rs, 4, 32);
      rs += __shfl_xor(rs, 8, 32);
      lrow[r] = lrow[r] * alpha + rs;
      o0[r] *= alpha; o1[r] *= alpha; o2[r] *= alpha; o3[r] *= alpha;

      const int row = r + hi8;                   // C-layout M for this lane
      ldsw[row * 32 + nl]      = to_bf16(pa);
      ldsw[row * 32 + 16 + nl] = to_bf16(pb);
    }
    // Same-wave LDS RAW: split-counter wait, no barrier needed.
    asm volatile("s_wait_dscnt 0" ::: "memory");

    const v16bf pf = load_fragA(ldsw, lane, 32);  // P as 16x32 A fragment
    o0 = wmma_bf16(pf, load_fragB(vptr + (size_t) 0 * S + kb0, lane, S), o0);
    o1 = wmma_bf16(pf, load_fragB(vptr + (size_t)16 * S + kb0, lane, S), o1);
    o2 = wmma_bf16(pf, load_fragB(vptr + (size_t)32 * S + kb0, lane, S), o2);
    o3 = wmma_bf16(pf, load_fragB(vptr + (size_t)48 * S + kb0, lane, S), o3);
  }

  const int h = bh & (H - 1);
  const int b = bh >> 4;
  const size_t rowbase = (size_t)b * S + qbase + hi8;
#pragma unroll
  for (int r = 0; r < 8; ++r) {
    const float inv = 1.0f / lrow[r];
    __bf16* dst = aout + (rowbase + r) * 1024 + h * 64 + nl;
    dst[0]  = to_bf16(o0[r] * inv);
    dst[16] = to_bf16(o1[r] * inv);
    dst[32] = to_bf16(o2[r] * inv);
    dst[48] = to_bf16(o3[r] * inv);
  }
}

// ---------------------------------------------------------------------------
// Launch
// ---------------------------------------------------------------------------
extern "C" void kernel_launch(void* const* d_in, const int* in_sizes, int n_in,
                              void* d_out, int out_size, void* d_ws, size_t ws_size,
                              hipStream_t stream) {
  (void)in_sizes; (void)n_in; (void)out_size; (void)ws_size;
  const float* x     = (const float*)d_in[0];
  const float* w_qkv = (const float*)d_in[1];
  const float* w_out = (const float*)d_in[2];
  const float* b_out = (const float*)d_in[3];
  float* out = (float*)d_out;

  constexpr int Bn = 2, S = 2048, D = 1024, H = 16;
  constexpr size_t M = (size_t)Bn * S;  // 4096 rows

  char* p = (char*)d_ws;
  __bf16* xb    = (__bf16*)p; p += M * D * sizeof(__bf16);             // 8 MB
  __bf16* wqkvb = (__bf16*)p; p += (size_t)3 * D * D * sizeof(__bf16); // 6 MB
  __bf16* woutb = (__bf16*)p; p += (size_t)D * D * sizeof(__bf16);     // 2 MB
  __bf16* qkvb  = (__bf16*)p; p += M * 3 * D * sizeof(__bf16);         // 24 MB
  __bf16* qb    = (__bf16*)p; p += M * D * sizeof(__bf16);             // 8 MB
  __bf16* kbuf  = (__bf16*)p; p += M * D * sizeof(__bf16);             // 8 MB
  __bf16* vbT   = (__bf16*)p; p += M * D * sizeof(__bf16);             // 8 MB
  __bf16* aout  = (__bf16*)p; p += M * D * sizeof(__bf16);             // 8 MB

  // 1) fp32 -> bf16
  {
    int n = (int)(M * D);
    f32_to_bf16_kernel<<<(n + 255) / 256, 256, 0, stream>>>(x, xb, n);
    n = 3 * D * D;
    f32_to_bf16_kernel<<<(n + 255) / 256, 256, 0, stream>>>(w_qkv, wqkvb, n);
    n = D * D;
    f32_to_bf16_kernel<<<(n + 255) / 256, 256, 0, stream>>>(w_out, woutb, n);
  }

  // 2) QKV GEMM: [4096,1024] x [3072,1024]^T -> bf16 [4096,3072]
  gemm_bf16_wmma<true><<<dim3((unsigned)(M / 32), (3 * D) / 256), 128, 0, stream>>>(
      xb, wqkvb, nullptr, qkvb, 3 * D, D);

  // 3) RoPE + repack (q,k rotated; v transposed to [bh, hd, S])
  {
    const int n = Bn * H * S * 32;
    rope_pack_kernel<<<n / 256, 256, 0, stream>>>(qkvb, qb, kbuf, vbT);
  }

  // 4) Flash attention -> bf16 [B*S, D]
  attn_kernel<<<dim3(S / 64, Bn * H), 128, 0, stream>>>(qb, kbuf, vbT, aout);

  // 5) Output projection: [4096,1024] x [1024,1024]^T + bias -> fp32 d_out
  gemm_bf16_wmma<false><<<dim3((unsigned)(M / 32), D / 256), 128, 0, stream>>>(
      aout, woutb, b_out, out, D, D);
}
